// TransformerEncoder_47132971106408
// MI455X (gfx1250) — compile-verified, matrix operand order checked by reading, not run on hardware
//
#include <hip/hip_runtime.h>
#include <hip/hip_bf16.h>

// ---------------------------------------------------------------------------
// Transformer encoder for MI455X (gfx1250): all GEMMs + attention through
// v_wmma_f32_16x16x32_bf16 (bf16 inputs, f32 accumulate), async global->LDS
// staging (global_load_async_to_lds_b128), global_prefetch for next tiles.
// B=2, S=2048, D=512, H=8, DH=64, NX=6, DF=2048. M = B*S = 4096.
// ---------------------------------------------------------------------------

#define B_    2
#define S_    2048
#define D_    512
#define H_    8
#define DH_   64
#define NX_   6
#define DF_   2048
#define M_    (B_ * S_)

typedef __attribute__((ext_vector_type(16))) __bf16 v16bf;
typedef __attribute__((ext_vector_type(8)))  float  v8f;

// --- gfx1250 async global->LDS path (compile-safe detection) ---------------
#if defined(__has_builtin)
#if __has_builtin(__builtin_amdgcn_global_load_async_to_lds_b128)
#define USE_ASYNC_LDS 1
#endif
#if __has_builtin(__builtin_amdgcn_s_wait_asynccnt)
#define HAVE_WAIT_ASYNC 1
#endif
#endif

#ifdef USE_ASYNC_LDS
// Builtin signature (from clang diagnostic):
//   (vector_size(16) int __device__*  /*AS1*/,  ... __shared__* /*AS3*/, Ii, Ii)
typedef int vi4 __attribute__((vector_size(16)));
typedef __attribute__((address_space(1))) vi4 as1_vi4;
typedef __attribute__((address_space(3))) vi4 as3_vi4;
static __device__ __forceinline__ void async_copy16(const void* g, void* l) {
    __builtin_amdgcn_global_load_async_to_lds_b128((as1_vi4*)g, (as3_vi4*)l, 0, 0);
}
static __device__ __forceinline__ void async_wait0() {
#ifdef HAVE_WAIT_ASYNC
    __builtin_amdgcn_s_wait_asynccnt(0);
#else
    asm volatile("s_wait_asynccnt 0" ::: "memory");
#endif
}
#else
static __device__ __forceinline__ void async_copy16(const void* g, void* l) {
    *(uint4*)l = *(const uint4*)g;
}
static __device__ __forceinline__ void async_wait0() {}
#endif

static __device__ __forceinline__ unsigned short f2bf(float f) {
    unsigned u = __float_as_uint(f);
    u = (u + 0x7FFFu + ((u >> 16) & 1u)) >> 16;   // round-to-nearest-even
    return (unsigned short)u;
}

static __device__ __forceinline__ v8f vzero8() {
    v8f z = {0.f, 0.f, 0.f, 0.f, 0.f, 0.f, 0.f, 0.f};
    return z;
}

// K-pair base for 16-bit A/B fragments (ISA 7.12.2):
// VGPR p<4: K = {2p,2p+1} (+8 for lanes 16-31); p>=4: K = 16+{...} (+8).
static __device__ __forceinline__ int kpair_base(int p, int khalf) {
    return (p < 4 ? 2 * p : 16 + 2 * (p - 4)) + 8 * khalf;
}

// Load a 16x32 bf16 A-fragment (or 32x16 B-fragment from an N-major tile).
// Rows indexed by lane&15, K by the pair mapping; pairs are contiguous so
// each VGPR is one 32-bit load (compiler merges them into ds_load_b128).
static __device__ __forceinline__ v16bf load_frag16(const __bf16* base, int stride, int lane) {
    const int row = lane & 15, khalf = lane >> 4;
    union { unsigned u[8]; v16bf v; } f;
    const __bf16* r = base + (size_t)row * stride;
#pragma unroll
    for (int p = 0; p < 8; ++p)
        f.u[p] = *(const unsigned*)(r + kpair_base(p, khalf));
    return f.v;
}

static __device__ __forceinline__ v8f wmma_bf16(v16bf a, v16bf b, v8f c) {
    return __builtin_amdgcn_wmma_f32_16x16x32_bf16(false, a, false, b, (short)0, c, false, false);
}

// ---------------------------------------------------------------------------
// Embedding + positional encoding (faithful to reference's exponent quirk).
// ---------------------------------------------------------------------------
__global__ __launch_bounds__(256) void embed_kernel(
    const int* __restrict__ x, const float* __restrict__ emb,
    float* __restrict__ Hf, __bf16* __restrict__ Hb)
{
    const int bs  = blockIdx.x;          // 0..M_-1
    const int pos = bs % S_;
    const int tok = x[bs];
#pragma unroll
    for (int i = 0; i < 2; ++i) {
        const int d = threadIdx.x + i * 256;
        const int j = d >> 1;
        const float e = (d & 1) ? (2.0f * (float)(2 * j + 1) / (float)D_)
                                : (2.0f * (float)(2 * j)     / (float)D_);
        const float ang = (float)pos * __powf(10000.0f, -e);
        const float pe  = (d & 1) ? __cosf(ang) : __sinf(ang);
        const float h   = emb[(size_t)tok * D_ + d] * 22.62741699796952f + pe; // sqrt(512)
        Hf[(size_t)bs * D_ + d] = h;
        ((unsigned short*)Hb)[(size_t)bs * D_ + d] = f2bf(h);
    }
}

// ---------------------------------------------------------------------------
// f32 -> bf16 conversion (weights), 4 elements / thread.
// ---------------------------------------------------------------------------
__global__ __launch_bounds__(256) void cvt_bf16_kernel(
    const float* __restrict__ src, unsigned short* __restrict__ dst, int n4)
{
    const int i = blockIdx.x * 256 + threadIdx.x;
    if (i < n4) {
        const float4 f = ((const float4*)src)[i];
        union { unsigned short s[4]; uint2 u; } o;
        o.s[0] = f2bf(f.x); o.s[1] = f2bf(f.y);
        o.s[2] = f2bf(f.z); o.s[3] = f2bf(f.w);
        ((uint2*)dst)[i] = o.u;
    }
}

// ---------------------------------------------------------------------------
// GEMM: C[MxN] = A[MxK](bf16) @ W[KxN](bf16) + bias (+ReLU) (+residual).
// Block tile 128x128, K-step 32. 8 waves: 4x2, each wave 32x64 = 2x4 WMMA.
// Compile-time variant flags keep the epilogue branch-free.
// ---------------------------------------------------------------------------
#define GBM 128
#define GBN 128
#define GBK 32
#define APAD 8   // +8 bf16 -> row stride 40 elems = 80B (16B-multiple)

template <int RELU, int RESID, int WF32, int WBF16>
__global__ __launch_bounds__(256) void gemm_bf16_kernel(
    const __bf16* __restrict__ A, const __bf16* __restrict__ W,
    const float* __restrict__ bias, const float* __restrict__ resid,
    float* __restrict__ Cf, __bf16* __restrict__ Cb,
    int M, int N, int K)
{
    __shared__ __align__(16) __bf16 As[GBM][GBK + APAD];   // [m][k]
    __shared__ __align__(16) __bf16 BsT[GBN][GBK + APAD];  // [n][k]  (N-major)

    const int tid = threadIdx.x;
    const int lane = tid & 31, wid = tid >> 5;
    const int wm = wid & 3, wn = wid >> 2;                 // 4x2 waves, 32x64 each
    const int bm = blockIdx.y * GBM, bn = blockIdx.x * GBN;

    v8f acc[2][4];
#pragma unroll
    for (int i = 0; i < 2; ++i)
#pragma unroll
        for (int j = 0; j < 4; ++j) acc[i][j] = vzero8();

    for (int k0 = 0; k0 < K; k0 += GBK) {
        // A tile: 128x32 bf16 = 512 x b128 chunks, 2/thread (async when avail).
#pragma unroll
        for (int c = 0; c < 2; ++c) {
            const int chunk = tid + c * 256;
            const int row = chunk >> 2;            // 4 chunks per row
            const int col = (chunk & 3) * 8;
            async_copy16(A + (size_t)(bm + row) * K + k0 + col, &As[row][col]);
        }
        // W tile: 32x128 bf16 = 512 x b128 chunks, 2/thread; transpose-store.
#pragma unroll
        for (int c = 0; c < 2; ++c) {
            const int chunk = tid + c * 256;
            const int row = chunk >> 4;            // 16 chunks per row
            const int col = (chunk & 15) * 8;
            const uint4 d = *(const uint4*)(W + (size_t)(k0 + row) * N + bn + col);
            const unsigned short* s = (const unsigned short*)&d;
#pragma unroll
            for (int i = 0; i < 8; ++i)
                *((unsigned short*)&BsT[col + i][row]) = s[i];
        }
        // Prefetch next K-tile into L2 while this one is consumed.
        if (k0 + GBK < K) {
            __builtin_prefetch(A + (size_t)(bm + (tid >> 1)) * K + k0 + GBK + (tid & 1) * 16, 0, 0);
            __builtin_prefetch(W + (size_t)(k0 + GBK + (tid >> 4)) * N + bn + (tid & 15) * 8, 0, 0);
        }
        async_wait0();
        __syncthreads();

        v16bf af[2], bf[4];
#pragma unroll
        for (int i = 0; i < 2; ++i)
            af[i] = load_frag16(&As[wm * 32 + i * 16][0], GBK + APAD, lane);
#pragma unroll
        for (int j = 0; j < 4; ++j)
            bf[j] = load_frag16(&BsT[wn * 64 + j * 16][0], GBK + APAD, lane);

#pragma unroll
        for (int i = 0; i < 2; ++i)
#pragma unroll
            for (int j = 0; j < 4; ++j)
                acc[i][j] = wmma_bf16(af[i], bf[j], acc[i][j]);
        __syncthreads();
    }

    // Epilogue. C layout: VGPR r -> row = r + 8*(lane>=16), col = lane&15.
    const int half = lane >> 4, nl = lane & 15;
#pragma unroll
    for (int j = 0; j < 4; ++j) {
        const int col = bn + wn * 64 + j * 16 + nl;
        const float bv = bias[col];
#pragma unroll
        for (int i = 0; i < 2; ++i) {
#pragma unroll
            for (int r = 0; r < 8; ++r) {
                const int row = bm + wm * 32 + i * 16 + r + 8 * half;
                float v = acc[i][j][r] + bv;
                if constexpr (RELU) v = v > 0.0f ? v : 0.0f;
                const size_t idx = (size_t)row * N + col;
                if constexpr (RESID) v += resid[idx];
                if constexpr (WF32)  Cf[idx] = v;
                if constexpr (WBF16) ((unsigned short*)Cb)[idx] = f2bf(v);
            }
        }
    }
}

// ---------------------------------------------------------------------------
// Flash attention. Block = 128 thr (4 waves) on one (b,h); each wave owns a
// 16-row query tile; keys streamed in blocks of 32 via shared LDS staging.
// ---------------------------------------------------------------------------
#define AKB 32

__global__ __launch_bounds__(128) void attn_kernel(
    const __bf16* __restrict__ Q, const __bf16* __restrict__ Km,
    const __bf16* __restrict__ V, __bf16* __restrict__ O)
{
    __shared__ __align__(16) __bf16 Ks[AKB][DH_ + 8];   // [key][dim]   (N-major for scores)
    __shared__ __align__(16) __bf16 Vt[DH_][AKB + 8];   // [dim][key]   (N-major for P@V)
    __shared__ __align__(16) __bf16 Pt[4][16][AKB];     // per-wave P tile [row][key]

    const int tid = threadIdx.x, lane = tid & 31, wid = tid >> 5;
    const int ngq = S_ / 64;
    const int qt4 = blockIdx.x % ngq;
    const int hh  = (blockIdx.x / ngq) % H_;
    const int bb  = blockIdx.x / (ngq * H_);
    const int q0  = qt4 * 64 + wid * 16;

    const size_t hoff = (size_t)hh * DH_;
    const __bf16* qbase = Q  + (size_t)(bb * S_) * D_ + hoff;
    const __bf16* kbase = Km + (size_t)(bb * S_) * D_ + hoff;
    const __bf16* vbase = V  + (size_t)(bb * S_) * D_ + hoff;

    // Q fragments held in registers for the whole loop (dims 0..31, 32..63).
    const v16bf a0 = load_frag16(qbase + (size_t)q0 * D_,      D_, lane);
    const v16bf a1 = load_frag16(qbase + (size_t)q0 * D_ + 32, D_, lane);

    v8f o0 = vzero8(), o1 = vzero8(), o2 = vzero8(), o3 = vzero8();
    float mrow[8], lrow[8];
#pragma unroll
    for (int r = 0; r < 8; ++r) { mrow[r] = -1e30f; lrow[r] = 0.0f; }
    const float scl = 0.125f;  // 1/sqrt(DH)

    const int half = lane >> 4, nl = lane & 15;

    for (int j0 = 0; j0 < S_; j0 += AKB) {
        // Stage K block (async, natural layout) and V block (transposed).
#pragma unroll
        for (int c = 0; c < 2; ++c) {
            const int chunk = tid + c * 128;          // 0..255, 8 chunks/key
            const int kr = chunk >> 3;
            const int kc = (chunk & 7) * 8;
            async_copy16(kbase + (size_t)(j0 + kr) * D_ + kc, &Ks[kr][kc]);
            const uint4 dv = *(const uint4*)(vbase + (size_t)(j0 + kr) * D_ + kc);
            const unsigned short* s = (const unsigned short*)&dv;
#pragma unroll
            for (int i = 0; i < 8; ++i)
                *((unsigned short*)&Vt[kc + i][kr]) = s[i];
        }
        async_wait0();
        __syncthreads();

        // Scores: S0 = keys j0..+15, S1 = keys j0+16..+31 (each 16x16, K=64).
        const v16bf bk00 = load_frag16(&Ks[0][0],   DH_ + 8, lane);
        const v16bf bk01 = load_frag16(&Ks[0][32],  DH_ + 8, lane);
        const v16bf bk10 = load_frag16(&Ks[16][0],  DH_ + 8, lane);
        const v16bf bk11 = load_frag16(&Ks[16][32], DH_ + 8, lane);
        v8f s0 = vzero8(), s1 = vzero8();
        s0 = wmma_bf16(a0, bk00, s0);  s0 = wmma_bf16(a1, bk01, s0);
        s1 = wmma_bf16(a0, bk10, s1);  s1 = wmma_bf16(a1, bk11, s1);

        // Online softmax, rows reduced across the 16 lanes of each half.
        float p0[8], p1[8];
#pragma unroll
        for (int r = 0; r < 8; ++r) {
            const float v0 = s0[r] * scl, v1 = s1[r] * scl;
            float mx = fmaxf(v0, v1);
#pragma unroll
            for (int off = 1; off < 16; off <<= 1)
                mx = fmaxf(mx, __shfl_xor(mx, off, 32));
            const float mnew = fmaxf(mrow[r], mx);
            const float corr = __expf(mrow[r] - mnew);
            const float e0 = __expf(v0 - mnew), e1 = __expf(v1 - mnew);
            float rs = e0 + e1;
#pragma unroll
            for (int off = 1; off < 16; off <<= 1)
                rs += __shfl_xor(rs, off, 32);
            lrow[r] = lrow[r] * corr + rs;
            mrow[r] = mnew;
            o0[r] *= corr; o1[r] *= corr; o2[r] *= corr; o3[r] *= corr;
            p0[r] = e0; p1[r] = e1;
        }

        // C-layout -> A-fragment layout via wave-private LDS round-trip.
#pragma unroll
        for (int r = 0; r < 8; ++r) {
            const int row = r + 8 * half;
            *((unsigned short*)&Pt[wid][row][nl])      = f2bf(p0[r]);
            *((unsigned short*)&Pt[wid][row][16 + nl]) = f2bf(p1[r]);
        }
        asm volatile("s_wait_dscnt 0" ::: "memory");   // wave-internal LDS RAW
        const v16bf ap = load_frag16(&Pt[wid][0][0], AKB, lane);

        // O += P (16x32) @ V (32x64): 4 WMMA over the dh column tiles.
        const v16bf bv0 = load_frag16(&Vt[0][0],  AKB + 8, lane);
        const v16bf bv1 = load_frag16(&Vt[16][0], AKB + 8, lane);
        const v16bf bv2 = load_frag16(&Vt[32][0], AKB + 8, lane);
        const v16bf bv3 = load_frag16(&Vt[48][0], AKB + 8, lane);
        o0 = wmma_bf16(ap, bv0, o0);
        o1 = wmma_bf16(ap, bv1, o1);
        o2 = wmma_bf16(ap, bv2, o2);
        o3 = wmma_bf16(ap, bv3, o3);
        __syncthreads();
    }

    // Normalize and scatter context (bf16) back into [B,S,D] head layout.
#pragma unroll
    for (int r = 0; r < 8; ++r) {
        const float inv = 1.0f / lrow[r];
        const int row = q0 + r + 8 * half;
        unsigned short* out = (unsigned short*)O + (size_t)(bb * S_ + row) * D_ + hoff + nl;
        out[0]  = f2bf(o0[r] * inv);
        out[16] = f2bf(o1[r] * inv);
        out[32] = f2bf(o2[r] * inv);
        out[48] = f2bf(o3[r] * inv);
    }
}

// ---------------------------------------------------------------------------
// LayerNorm over D=512 (residual already folded in by the GEMM epilogue).
// ---------------------------------------------------------------------------
template <int WB>
__global__ __launch_bounds__(128) void ln_kernel(
    const float* __restrict__ X, const float* __restrict__ g,
    const float* __restrict__ b, float* __restrict__ Yf, __bf16* __restrict__ Yb)
{
    __shared__ float red[2][4];
    const int row = blockIdx.x, tid = threadIdx.x;
    const int lane = tid & 31, wid = tid >> 5;
    const float* x = X + (size_t)row * D_;

    float v[4], s = 0.0f, sq = 0.0f;
#pragma unroll
    for (int i = 0; i < 4; ++i) {
        v[i] = x[tid + i * 128];
        s += v[i]; sq += v[i] * v[i];
    }
#pragma unroll
    for (int off = 1; off < 32; off <<= 1) {
        s  += __shfl_xor(s,  off, 32);
        sq += __shfl_xor(sq, off, 32);
    }
    if (lane == 0) { red[0][wid] = s; red[1][wid] = sq; }
    __syncthreads();
    s  = red[0][0] + red[0][1] + red[0][2] + red[0][3];
    sq = red[1][0] + red[1][1] + red[1][2] + red[1][3];

    const float mu   = s * (1.0f / D_);
    const float var  = sq * (1.0f / D_) - mu * mu;
    const float rstd = rsqrtf(var + 1e-5f);
#pragma unroll
    for (int i = 0; i < 4; ++i) {
        const int d = tid + i * 128;
        const float y = (v[i] - mu) * rstd * g[d] + b[d];
        Yf[(size_t)row * D_ + d] = y;
        if constexpr (WB) ((unsigned short*)Yb)[(size_t)row * D_ + d] = f2bf(y);
    }
}

// ---------------------------------------------------------------------------
// Host orchestration. Workspace layout (bytes, 256-aligned), ~61 MB total.
// ---------------------------------------------------------------------------
extern "C" void kernel_launch(void* const* d_in, const int* in_sizes, int n_in,
                              void* d_out, int out_size, void* d_ws, size_t ws_size,
                              hipStream_t stream) {
    (void)in_sizes; (void)n_in; (void)out_size; (void)ws_size;

    const int*   x   = (const int*)  d_in[0];
    const float* emb = (const float*)d_in[1];
    const float* Wq = (const float*)d_in[2],  *bq  = (const float*)d_in[3];
    const float* Wk = (const float*)d_in[4],  *bk  = (const float*)d_in[5];
    const float* Wv = (const float*)d_in[6],  *bv  = (const float*)d_in[7];
    const float* Wo = (const float*)d_in[8],  *bo  = (const float*)d_in[9];
    const float* W1 = (const float*)d_in[10], *b1  = (const float*)d_in[11];
    const float* W2 = (const float*)d_in[12], *b2  = (const float*)d_in[13];
    const float* g1 = (const float*)d_in[14], *be1 = (const float*)d_in[15];
    const float* g2 = (const float*)d_in[16], *be2 = (const float*)d_in[17];

    char* ws = (char*)d_ws;
    size_t off = 0;
    auto carve = [&](size_t bytes) { char* p = ws + off; off += (bytes + 255) & ~(size_t)255; return p; };

    float*  h    = (float*) carve((size_t)M_ * D_  * 4);   // f32 activations
    __bf16* hb   = (__bf16*)carve((size_t)M_ * D_  * 2);   // bf16 activations
    __bf16* qb   = (__bf16*)carve((size_t)M_ * D_  * 2);
    __bf16* kb   = (__bf16*)carve((size_t)M_ * D_  * 2);
    __bf16* vb   = (__bf16*)carve((size_t)M_ * D_  * 2);
    __bf16* ctxb = (__bf16*)carve((size_t)M_ * D_  * 2);
    float*  tmp  = (float*) carve((size_t)M_ * D_  * 4);   // pre-LN (resid folded)
    __bf16* ff1b = (__bf16*)carve((size_t)M_ * DF_ * 2);
    __bf16* wqb  = (__bf16*)carve((size_t)D_ * D_  * 2);   // per-layer bf16 weights
    __bf16* wkb  = (__bf16*)carve((size_t)D_ * D_  * 2);
    __bf16* wvb  = (__bf16*)carve((size_t)D_ * D_  * 2);
    __bf16* wob  = (__bf16*)carve((size_t)D_ * D_  * 2);
    __bf16* w1b  = (__bf16*)carve((size_t)D_ * DF_ * 2);
    __bf16* w2b  = (__bf16*)carve((size_t)DF_ * D_ * 2);

    auto cvt = [&](const float* src, __bf16* dst, int n) {
        const int n4 = n / 4;
        cvt_bf16_kernel<<<(n4 + 255) / 256, 256, 0, stream>>>(src, (unsigned short*)dst, n4);
    };

    // h = emb[x]*sqrt(D) + PE  (f32 + bf16)
    embed_kernel<<<M_, 256, 0, stream>>>(x, emb, h, hb);

    for (int l = 0; l < NX_; ++l) {
        const size_t wdd = (size_t)l * D_ * D_;
        const size_t wdf = (size_t)l * D_ * DF_;
        cvt(Wq + wdd, wqb, D_ * D_);
        cvt(Wk + wdd, wkb, D_ * D_);
        cvt(Wv + wdd, wvb, D_ * D_);
        cvt(Wo + wdd, wob, D_ * D_);
        cvt(W1 + wdf, w1b, D_ * DF_);
        cvt(W2 + wdf, w2b, DF_ * D_);

        const dim3 gD (D_  / GBN, M_ / GBM);   // N=512
        const dim3 gDF(DF_ / GBN, M_ / GBM);   // N=2048

        // Q/K/V projections (bf16 out for attention).
        gemm_bf16_kernel<0,0,0,1><<<gD, 256, 0, stream>>>(hb, wqb, bq + (size_t)l * D_,
                                                          nullptr, nullptr, qb, M_, D_, D_);
        gemm_bf16_kernel<0,0,0,1><<<gD, 256, 0, stream>>>(hb, wkb, bk + (size_t)l * D_,
                                                          nullptr, nullptr, kb, M_, D_, D_);
        gemm_bf16_kernel<0,0,0,1><<<gD, 256, 0, stream>>>(hb, wvb, bv + (size_t)l * D_,
                                                          nullptr, nullptr, vb, M_, D_, D_);

        // Flash attention -> ctx (bf16).
        attn_kernel<<<B_ * H_ * (S_ / 64), 128, 0, stream>>>(qb, kb, vb, ctxb);

        // Output projection + residual; then LN1.
        gemm_bf16_kernel<0,1,1,0><<<gD, 256, 0, stream>>>(ctxb, wob, bo + (size_t)l * D_,
                                                          h, tmp, nullptr, M_, D_, D_);
        ln_kernel<1><<<M_, 128, 0, stream>>>(tmp, g1 + (size_t)l * D_, be1 + (size_t)l * D_, h, hb);

        // FFN: ReLU(h@W1+b1)@W2+b2 + residual; then LN2.
        gemm_bf16_kernel<1,0,0,1><<<gDF, 256, 0, stream>>>(hb, w1b, b1 + (size_t)l * DF_,
                                                           nullptr, nullptr, ff1b, M_, DF_, D_);
        gemm_bf16_kernel<0,1,1,0><<<gD, 256, 0, stream>>>(ff1b, w2b, b2 + (size_t)l * D_,
                                                          h, tmp, nullptr, M_, D_, DF_);

        const bool last = (l == NX_ - 1);
        if (last) {
            ln_kernel<0><<<M_, 128, 0, stream>>>(tmp, g2 + (size_t)l * D_, be2 + (size_t)l * D_,
                                                 (float*)d_out, nullptr);
        } else {
            ln_kernel<1><<<M_, 128, 0, stream>>>(tmp, g2 + (size_t)l * D_, be2 + (size_t)l * D_,
                                                 h, hb);
        }
    }
}
